// LightMamba_23828478558712
// MI455X (gfx1250) — matrix operand order repaired and checked by reading, not compiled, not run on hardware
//
#include <hip/hip_runtime.h>
#include <hip/hip_bf16.h>
#include <math.h>

// ---- model dims ----
#define BB 16
#define TT 512
#define SEG 32
#define CD 256
#define DI 512
#define DSN 16
#define MROWS (BB*TT)          // 8192

typedef __attribute__((ext_vector_type(16))) __bf16 v16bf;
typedef __attribute__((ext_vector_type(8)))  float  v8f;
typedef __attribute__((ext_vector_type(4)))  unsigned int v4u;
typedef __attribute__((ext_vector_type(8)))  int v8i_;
typedef __attribute__((ext_vector_type(4)))  int v4i_;

#define HAS_TDM __has_builtin(__builtin_amdgcn_tensor_load_to_lds)

__device__ __forceinline__ float geluf(float x) {
    return 0.5f * x * (1.0f + erff(x * 0.70710678118654752440f));
}
__device__ __forceinline__ float siluf(float x) {
    return x / (1.0f + __expf(-x));
}
__device__ __forceinline__ float softplusf(float x) {
    return (x > 20.0f) ? x : log1pf(__expf(x));
}

// ---- TDM: stage a (rows x rowlen) fp32 tile (row-major, leading dim src_ld_elems)
// from global into LDS at byte offset lds_off. D# layout per CDNA5 ISA §8.3/8.4.
// This toolchain's builtin is the 6-arg form:
//   (uint32x4 g0, int32x8 g1, int32x4 g2, int32x4 g3, int32x8 pad, i32 cpol)
__device__ __forceinline__ void tdm_load_tile_f32(unsigned lds_off, const void* gsrc,
                                                  unsigned rows, unsigned rowlen,
                                                  unsigned src_ld_elems) {
#if HAS_TDM
    unsigned long long ga = (unsigned long long)(size_t)gsrc;
    v4u g0;
    g0[0] = 1u;                                    // count=1 (valid user D#), no gather
    g0[1] = lds_off;                               // lds_addr [63:32]
    g0[2] = (unsigned)ga;                          // global_addr[31:0]
    g0[3] = (unsigned)((ga >> 32) & 0x01FFFFFFu)   // global_addr[56:32]
            | (2u << 30);                          // type = 2 ("image")
    v8i_ g1;
    g1[0] = (int)(2u << 16);                       // wg_mask=0, data_size=2 -> 4 bytes
    g1[1] = (int)((rowlen & 0xFFFFu) << 16);       // tensor_dim0[15:0] @ bits 63:48
    g1[2] = (int)(((rowlen >> 16) & 0xFFFFu)       // tensor_dim0[31:16]
            | ((rows & 0xFFFFu) << 16));           // tensor_dim1[15:0]
    g1[3] = (int)(((rows >> 16) & 0xFFFFu)         // tensor_dim1[31:16]
            | ((rowlen & 0xFFFFu) << 16));         // tile_dim0 = rowlen
    g1[4] = (int)(rows & 0xFFFFu);                 // tile_dim1 = rows, tile_dim2 = 0
    g1[5] = (int)src_ld_elems;                     // tensor_dim0_stride[31:0]
    g1[6] = 0;                                     // stride hi / dim1_stride (unused, 2D)
    g1[7] = 0;
    v4i_ z4 = {0, 0, 0, 0};                        // groups 2/3 unused for 2D tensors
    v8i_ z8 = {0, 0, 0, 0, 0, 0, 0, 0};
    __builtin_amdgcn_tensor_load_to_lds(g0, g1, z4, z4, z8, 0);
    __builtin_amdgcn_s_wait_tensorcnt(0);
#endif
}

// A fragment: 16x32 bf16 tile from row-major fp32 (leading dim ld).
// lane L: M = L%16 ; elements 0..15 -> K = 16*(L/16) + 0..15
__device__ __forceinline__ v16bf load_a_frag(const float* __restrict__ base, int ld) {
    int lane = threadIdx.x & 31;
    const float* p = base + (size_t)(lane & 15) * ld + ((lane >> 4) << 4);
    v16bf v;
#pragma unroll
    for (int i = 0; i < 8; ++i) v[i] = (__bf16)p[i];
#pragma unroll
    for (int i = 0; i < 8; ++i) v[i + 8] = (__bf16)p[8 + i];
    return v;
}

// B fragment: 32x16 bf16 from weights stored (N,K) row-major (i.e. W for x@W.T).
// lane L: N = L%16 ; elements 0..15 -> K = 16*(L/16) + 0..15
__device__ __forceinline__ v16bf load_b_frag(const float* __restrict__ base, int ld) {
    int lane = threadIdx.x & 31;
    const float* p = base + (size_t)(lane & 15) * ld + ((lane >> 4) << 4);
    v16bf v;
#pragma unroll
    for (int i = 0; i < 16; ++i) v[i] = (__bf16)p[i];
    return v;
}

// ---------------- K0: aux head : haux = gelu(LN(aux @ aux_proj_w.T)) ----------------
__global__ void k_aux(const float* __restrict__ aux, const float* __restrict__ W,
                      const float* __restrict__ g, const float* __restrict__ b,
                      float* __restrict__ haux) {
    __shared__ float sa[CD];
    __shared__ float smean, srstd;
    int tid = threadIdx.x;
    for (int r = 0; r < BB; ++r) {
        float v = 0.f;
#pragma unroll
        for (int k = 0; k < 14; ++k) v += aux[r * 14 + k] * W[tid * 14 + k];
        sa[tid] = v;
        __syncthreads();
        if (tid == 0) {
            float s = 0.f, sq = 0.f;
            for (int i = 0; i < CD; ++i) { s += sa[i]; sq += sa[i] * sa[i]; }
            float m = s / (float)CD;
            smean = m;
            srstd = rsqrtf(sq / (float)CD - m * m + 1e-5f);
        }
        __syncthreads();
        haux[r * CD + tid] = geluf((v - smean) * srstd * g[tid] + b[tid]);
        __syncthreads();
    }
}

// ---------------- K1: h = gelu(LN(x_seg @ cir_proj_w.T)) + haux ----------------
__global__ void k_cir(const float* __restrict__ cir, const float* __restrict__ Wc,
                      const float* __restrict__ g, const float* __restrict__ b,
                      const float* __restrict__ haux, float* __restrict__ h) {
    __shared__ float sh[16][CD];
    __shared__ float smean[16], srstd[16];
    int tid  = threadIdx.x;
    int lane = tid & 31;
    int w    = tid >> 5;
    int m0   = blockIdx.x * 16;
    int bidx = m0 >> 9;   // row/T

    v16bf a = load_a_frag(cir + (size_t)m0 * SEG, SEG);   // K = SEG = 32, one step
#pragma unroll
    for (int j = 0; j < 2; ++j) {
        int n0 = w * 16 + j * 128;
        v16bf bf = load_b_frag(Wc + (size_t)n0 * SEG, SEG);
        v8f c = {};
        c = __builtin_amdgcn_wmma_f32_16x16x32_bf16(false, a, false, bf, (short)0, c, false, false);
        int col = n0 + (lane & 15);
        int rb  = (lane >> 4) << 3;
#pragma unroll
        for (int r = 0; r < 8; ++r) sh[rb + r][col] = c[r];
    }
    __syncthreads();
#pragma unroll
    for (int rr = 0; rr < 2; ++rr) {
        int r = w * 2 + rr;
        float s = 0.f, sq = 0.f;
#pragma unroll
        for (int i = 0; i < 8; ++i) { float v = sh[r][lane + 32 * i]; s += v; sq += v * v; }
#pragma unroll
        for (int off = 16; off > 0; off >>= 1) {
            s += __shfl_xor(s, off, 32);
            sq += __shfl_xor(sq, off, 32);
        }
        if (lane == 0) {
            float m = s / (float)CD;
            smean[r] = m;
            srstd[r] = rsqrtf(sq / (float)CD - m * m + 1e-5f);
        }
    }
    __syncthreads();
    for (int idx = tid; idx < 16 * CD; idx += 256) {
        int r = idx >> 8, cc = idx & 255;
        float v = (sh[r][cc] - smean[r]) * srstd[r] * g[cc] + b[cc];
        h[(size_t)(m0 + r) * CD + cc] = geluf(v) + haux[bidx * CD + cc];
    }
}

// ---------------- K2: row layernorm (m_norm), one wave per 256-col row ----------------
__global__ void k_rowln(const float* __restrict__ x, const float* __restrict__ g,
                        const float* __restrict__ b, float* __restrict__ out) {
    int wid  = (blockIdx.x * blockDim.x + threadIdx.x) >> 5;
    int lane = threadIdx.x & 31;
    const float* row = x + (size_t)wid * CD;
    float v[8], s = 0.f, sq = 0.f;
#pragma unroll
    for (int i = 0; i < 8; ++i) { v[i] = row[lane + 32 * i]; s += v[i]; sq += v[i] * v[i]; }
#pragma unroll
    for (int off = 16; off > 0; off >>= 1) {
        s += __shfl_xor(s, off, 32);
        sq += __shfl_xor(sq, off, 32);
    }
    float m = s / (float)CD;
    float rstd = rsqrtf(sq / (float)CD - m * m + 1e-5f);
    float* orow = out + (size_t)wid * CD;
#pragma unroll
    for (int i = 0; i < 8; ++i) { int cc = lane + 32 * i; orow[cc] = (v[i] - m) * rstd * g[cc] + b[cc]; }
}

// ---------------- K3: xz = hn @ in_proj_w.T  (M=8192, N=1024, K=256) ----------------
// A tile (16x256 fp32) staged to LDS by the Tensor Data Mover, shared by 8 waves.
__global__ void k_inproj(const float* __restrict__ A, const float* __restrict__ W,
                         float* __restrict__ C) {
    __shared__ float sA[16 * CD];
    int w    = threadIdx.x >> 5;
    int lane = threadIdx.x & 31;
    int m0   = blockIdx.x * 16;
    int n0   = (blockIdx.y * 8 + w) * 16;
#if HAS_TDM
    if (threadIdx.x < 32)
        tdm_load_tile_f32((unsigned)(size_t)sA, A + (size_t)m0 * CD, 16, CD, CD);
#else
    for (int i = threadIdx.x; i < 16 * CD; i += 256) sA[i] = A[(size_t)m0 * CD + i];
#endif
    __syncthreads();
    v8f c = {};
#pragma unroll
    for (int k0 = 0; k0 < CD; k0 += 32) {
        v16bf a = load_a_frag(sA + k0, CD);
        v16bf b = load_b_frag(W + (size_t)n0 * CD + k0, CD);
        c = __builtin_amdgcn_wmma_f32_16x16x32_bf16(false, a, false, b, (short)0, c, false, false);
    }
    int col = n0 + (lane & 15);
    int rb  = (lane >> 4) << 3;
#pragma unroll
    for (int r = 0; r < 8; ++r) C[(size_t)(m0 + rb + r) * (2 * DI) + col] = c[r];
}

// ---------------- K4: causal depthwise conv (DCONV=4) + silu ----------------
__global__ void k_conv(const float* __restrict__ xz, const float* __restrict__ cw,
                       const float* __restrict__ cb, float* __restrict__ xc) {
    int idx = blockIdx.x * blockDim.x + threadIdx.x;
    if (idx >= MROWS * DI) return;
    int d   = idx & (DI - 1);
    int row = idx >> 9;
    int t   = row & (TT - 1);
    int b   = row >> 9;
    float acc = cb[d];
#pragma unroll
    for (int k = 0; k < 4; ++k) {
        int tt = t - 3 + k;
        if (tt >= 0) acc += xz[(size_t)(b * TT + tt) * (2 * DI) + d] * cw[d * 4 + k];
    }
    xc[idx] = siluf(acc);
}

// ---------------- K5: x_proj (N=33) as one wave per (row, j) dot over K=512 ----------------
__global__ void k_xproj(const float* __restrict__ xc, const float* __restrict__ Wx,
                        float* __restrict__ dtr, float* __restrict__ Bm,
                        float* __restrict__ Cm) {
    int wid  = (blockIdx.x * blockDim.x + threadIdx.x) >> 5;
    int lane = threadIdx.x & 31;
    if (wid >= MROWS * 33) return;
    int row = wid / 33;
    int j   = wid - row * 33;
    const float* xr = xc + (size_t)row * DI;
    const float* wr = Wx + (size_t)j * DI;
    float s = 0.f;
#pragma unroll
    for (int i = 0; i < 16; ++i) s += xr[lane + 32 * i] * wr[lane + 32 * i];
#pragma unroll
    for (int off = 16; off > 0; off >>= 1) s += __shfl_xor(s, off, 32);
    if (lane == 0) {
        if (j == 0)        dtr[row] = s;
        else if (j <= DSN) Bm[row * DSN + (j - 1)] = s;
        else               Cm[row * DSN + (j - 1 - DSN)] = s;
    }
}

// ---------------- K6: selective scan. wave = (b, 2 channels); lanes = 16 states x 2 ----------------
__global__ void k_scan(const float* __restrict__ dtr, const float* __restrict__ Bm,
                       const float* __restrict__ Cm, const float* __restrict__ xc,
                       const float* __restrict__ Alog, const float* __restrict__ dtw,
                       const float* __restrict__ dtb, float* __restrict__ y) {
    int wid  = (blockIdx.x * blockDim.x + threadIdx.x) >> 5;   // 0..4095
    int lane = threadIdx.x & 31;
    int b     = wid >> 8;
    int dpair = wid & 255;
    int d = dpair * 2 + (lane >> 4);
    int s = lane & 15;
    float A  = -__expf(Alog[d * DSN + s]);
    float wd = dtw[d];
    float bd = dtb[d];
    float hst = 0.f;
    for (int t = 0; t < TT; ++t) {
        int row = b * TT + t;
        if (t + 1 < TT) {   // hide next step's loads behind the exp/fma chain
            __builtin_prefetch(&Bm[(row + 1) * DSN], 0, 0);
            __builtin_prefetch(&xc[(size_t)(row + 1) * DI + d], 0, 0);
        }
        float delta = softplusf(dtr[row] * wd + bd);
        float xcv = xc[(size_t)row * DI + d];
        float Bs  = Bm[row * DSN + s];
        float Cs  = Cm[row * DSN + s];
        hst = __expf(delta * A) * hst + delta * Bs * xcv;
        float p = hst * Cs;
        p += __shfl_xor(p, 1, 32);
        p += __shfl_xor(p, 2, 32);
        p += __shfl_xor(p, 4, 32);
        p += __shfl_xor(p, 8, 32);
        if (s == 0) y[(size_t)row * DI + d] = p;
    }
}

// ---------------- K7: y = (y + xc*D) * silu(z) ----------------
__global__ void k_gate(const float* __restrict__ xz, const float* __restrict__ xc,
                       const float* __restrict__ Dv, float* __restrict__ y) {
    int idx = blockIdx.x * blockDim.x + threadIdx.x;
    if (idx >= MROWS * DI) return;
    int d   = idx & (DI - 1);
    int row = idx >> 9;
    float yv = y[idx] + xc[idx] * Dv[d];
    float zv = xz[(size_t)row * (2 * DI) + DI + d];
    y[idx] = yv * siluf(zv);
}

// ---------------- K8: hfin = h + y @ out_proj_w.T  (M=8192, N=256, K=512) ----------------
// A tile (16x512 fp32, 32KB) staged to LDS by the Tensor Data Mover.
__global__ void k_outproj(const float* __restrict__ Y, const float* __restrict__ W,
                          const float* __restrict__ H, float* __restrict__ Hf) {
    __shared__ float sA[16 * DI];
    int w    = threadIdx.x >> 5;
    int lane = threadIdx.x & 31;
    int m0   = blockIdx.x * 16;
    int n0   = (blockIdx.y * 8 + w) * 16;
#if HAS_TDM
    if (threadIdx.x < 32)
        tdm_load_tile_f32((unsigned)(size_t)sA, Y + (size_t)m0 * DI, 16, DI, DI);
#else
    for (int i = threadIdx.x; i < 16 * DI; i += 256) sA[i] = Y[(size_t)m0 * DI + i];
#endif
    __syncthreads();
    v8f c = {};
#pragma unroll
    for (int k0 = 0; k0 < DI; k0 += 32) {
        v16bf a = load_a_frag(sA + k0, DI);
        v16bf b = load_b_frag(W + (size_t)n0 * DI + k0, DI);
        c = __builtin_amdgcn_wmma_f32_16x16x32_bf16(false, a, false, b, (short)0, c, false, false);
    }
    int col = n0 + (lane & 15);
    int rb  = (lane >> 4) << 3;
#pragma unroll
    for (int r = 0; r < 8; ++r) {
        size_t idx = (size_t)(m0 + rb + r) * CD + col;
        Hf[idx] = H[idx] + c[r];
    }
}

// ---------------- K9: hp = mean over T ----------------
__global__ void k_mean(const float* __restrict__ Hf, float* __restrict__ hp) {
    int tid = blockIdx.x * blockDim.x + threadIdx.x;
    if (tid >= BB * CD) return;
    int b = tid >> 8, cc = tid & 255;
    float s = 0.f;
    for (int t = 0; t < TT; ++t) s += Hf[(size_t)(b * TT + t) * CD + cc];
    hp[tid] = s * (1.0f / (float)TT);
}

// ---------------- K10: MLP head, single block ----------------
__global__ void k_head(const float* __restrict__ hp, const float* __restrict__ midw,
                       const float* __restrict__ midb, const float* __restrict__ mg,
                       const float* __restrict__ mb, const float* __restrict__ bng,
                       const float* __restrict__ bnb, const float* __restrict__ f1w,
                       const float* __restrict__ f1b, const float* __restrict__ f2w,
                       const float* __restrict__ f2b, float* __restrict__ out) {
    __shared__ float t1[CD];
    __shared__ float hbn[CD];
    __shared__ float f1[CD / 2];
    __shared__ float smean, srstd;
    int tid = threadIdx.x;
    const float bnscale = rsqrtf(1.0f + 1e-5f);
    for (int r = 0; r < BB; ++r) {
        float v = midb[tid];
        for (int k = 0; k < CD; ++k) v += hp[r * CD + k] * midw[tid * CD + k];
        t1[tid] = v;
        __syncthreads();
        if (tid == 0) {
            float s = 0.f, sq = 0.f;
            for (int i = 0; i < CD; ++i) { s += t1[i]; sq += t1[i] * t1[i]; }
            float m = s / (float)CD;
            smean = m;
            srstd = rsqrtf(sq / (float)CD - m * m + 1e-5f);
        }
        __syncthreads();
        float h2 = geluf((v - smean) * srstd * mg[tid] + mb[tid]);
        hbn[tid] = h2 * bnscale * bng[tid] + bnb[tid];
        __syncthreads();
        if (tid < CD / 2) {
            float a = f1b[tid];
            for (int k = 0; k < CD; ++k) a += hbn[k] * f1w[tid * CD + k];
            f1[tid] = geluf(a);
        }
        __syncthreads();
        if (tid == 0) {
            float o = f2b[0];
            for (int k = 0; k < CD / 2; ++k) o += f1[k] * f2w[k];
            out[r] = o;
        }
        __syncthreads();
    }
}

extern "C" void kernel_launch(void* const* d_in, const int* in_sizes, int n_in,
                              void* d_out, int out_size, void* d_ws, size_t ws_size,
                              hipStream_t stream) {
    const float* cir    = (const float*)d_in[0];
    const float* aux    = (const float*)d_in[1];
    const float* cirW   = (const float*)d_in[2];
    const float* cirG   = (const float*)d_in[3];
    const float* cirB   = (const float*)d_in[4];
    const float* auxW   = (const float*)d_in[5];
    const float* auxG   = (const float*)d_in[6];
    const float* auxB   = (const float*)d_in[7];
    const float* mG     = (const float*)d_in[8];
    const float* mB     = (const float*)d_in[9];
    const float* inW    = (const float*)d_in[10];
    const float* convW  = (const float*)d_in[11];
    const float* convB  = (const float*)d_in[12];
    const float* xpW    = (const float*)d_in[13];
    const float* dtW    = (const float*)d_in[14];
    const float* dtB    = (const float*)d_in[15];
    const float* Alog   = (const float*)d_in[16];
    const float* Dv     = (const float*)d_in[17];
    const float* outW   = (const float*)d_in[18];
    const float* midW   = (const float*)d_in[19];
    const float* midB   = (const float*)d_in[20];
    const float* midG   = (const float*)d_in[21];
    const float* midB2  = (const float*)d_in[22];
    const float* bnG    = (const float*)d_in[23];
    const float* bnB    = (const float*)d_in[24];
    const float* f1W    = (const float*)d_in[25];
    const float* f1B    = (const float*)d_in[26];
    const float* f2W    = (const float*)d_in[27];
    const float* f2B    = (const float*)d_in[28];
    float* out = (float*)d_out;

    // workspace layout (floats)
    float* ws    = (float*)d_ws;
    float* haux  = ws;                                  // 16*256
    float* h     = haux + BB * CD;                      // 8192*256
    float* hn    = h + (size_t)MROWS * CD;              // 8192*256 (reused as hfin)
    float* xz    = hn + (size_t)MROWS * CD;             // 8192*1024
    float* xconv = xz + (size_t)MROWS * 2 * DI;         // 8192*512
    float* dtr   = xconv + (size_t)MROWS * DI;          // 8192
    float* Bm    = dtr + MROWS;                         // 8192*16
    float* Cm    = Bm + (size_t)MROWS * DSN;            // 8192*16
    float* yb    = Cm + (size_t)MROWS * DSN;            // 8192*512
    float* hp    = yb + (size_t)MROWS * DI;             // 16*256
    float* hfin  = hn;                                  // hn dead after in_proj

    k_aux<<<1, 256, 0, stream>>>(aux, auxW, auxG, auxB, haux);
    k_cir<<<MROWS / 16, 256, 0, stream>>>(cir, cirW, cirG, cirB, haux, h);
    k_rowln<<<MROWS / 8, 256, 0, stream>>>(h, mG, mB, hn);
    k_inproj<<<dim3(MROWS / 16, 8), 256, 0, stream>>>(hn, inW, xz);
    k_conv<<<(MROWS * DI) / 256, 256, 0, stream>>>(xz, convW, convB, xconv);
    {
        int waves = MROWS * 33;
        int blocks = (waves * 32 + 255) / 256;
        k_xproj<<<blocks, 256, 0, stream>>>(xconv, xpW, dtr, Bm, Cm);
    }
    k_scan<<<(BB * (DI / 2) * 32) / 256, 256, 0, stream>>>(dtr, Bm, Cm, xconv,
                                                          Alog, dtW, dtB, yb);
    k_gate<<<(MROWS * DI) / 256, 256, 0, stream>>>(xz, xconv, Dv, yb);
    k_outproj<<<dim3(MROWS / 16, 2), 256, 0, stream>>>(yb, outW, h, hfin);
    k_mean<<<(BB * CD) / 256, 256, 0, stream>>>(hfin, hp);
    k_head<<<1, 256, 0, stream>>>(hp, midW, midB, midG, midB2, bnG, bnB,
                                  f1W, f1B, f2W, f2B, out);
}
